// ManifoldConstrainedHyperConnection_13649406066737
// MI455X (gfx1250) — compile-verified, matrix operand order checked by reading, not run on hardware
//
#include <hip/hip_runtime.h>
#include <hip/hip_bf16.h>
#include <math.h>

// Problem constants (from reference setup_inputs)
#define BT     8192          // B*T = 2*4096
#define WS     4             // streams
#define DD     2048          // feature dim
#define NOUT   24            // W*W + 2W
#define TOK_TILE 16          // tokens per workgroup == WMMA M
#define D_HALF 1024          // process D in two halves to bound LDS
#define PITCH  (D_HALF + 16) // bf16 elements per LDS row (2080B, 16B-aligned rows)
#define NCHUNK (DD / 32)     // 64 K-chunks of 32

typedef __attribute__((ext_vector_type(16))) __bf16 v16bf;
typedef __attribute__((ext_vector_type(8)))  __bf16 v8bf;
typedef __attribute__((ext_vector_type(8)))  float  v8f;
typedef __attribute__((ext_vector_type(4)))  float  v4f;

static __device__ inline unsigned short f32_to_bf16_bits(float f) {
    unsigned int u = __float_as_uint(f);
    unsigned int lsb = (u >> 16) & 1u;
    u += 0x7fffu + lsb;                 // round-to-nearest-even
    return (unsigned short)(u >> 16);
}
static __device__ inline __bf16 bf16_from_bits(unsigned short s) {
    union { unsigned short u; __bf16 b; } v; v.u = s; return v.b;
}

// ---------------------------------------------------------------------------
// One-shot: pack W_router (24 x 2048 f32) into WMMA B-fragment layout, bf16.
// bfrag[(kc*2 + tile)*32 + lane] = v16bf of B[k][n] for that lane.
// ISA 7.12.2: lane group g = lane>>4; element e holds K = kc*32 +
// (e>=8 ? 16 : 0) + g*8 + (e&7); column n = tile*16 + (lane&15).
// Padding columns (n >= 24) are stored as zero.
// ---------------------------------------------------------------------------
__global__ __launch_bounds__(64)
void pack_router_kernel(const float* __restrict__ Wr, v16bf* __restrict__ bfrag)
{
    const int lane = threadIdx.x & 31;
    const int tile = threadIdx.x >> 5;   // 0 or 1
    const int kc   = blockIdx.x;         // 0..63
    const int g    = lane >> 4;
    const int nc   = lane & 15;
    const int n    = tile * 16 + nc;

    v16bf v;
    #pragma unroll
    for (int e = 0; e < 16; ++e) {
        int k = kc * 32 + ((e >= 8) ? 16 : 0) + g * 8 + (e & 7);
        float w = (n < NOUT) ? Wr[n * DD + k] : 0.0f;
        v[e] = bf16_from_bits(f32_to_bf16_bits(w));
    }
    bfrag[(kc * 2 + tile) * 32 + lane] = v;
}

// ---------------------------------------------------------------------------
// Fused kernel: 16 tokens per workgroup.
//  P1: stream means of 4 streams into LDS as bf16 (A-matrix rows), RT loads
//      + distance-2 prefetch (HBM first touch)
//  P2: wave0 does (16 x 2048)(2048 x 24) router matmul via WMMA bf16
//  P3: per-token softmaxes + Sinkhorn -> fused 4x4 coefficient matrix
//  P4: stream x again in reverse (most-recent d first, NT last-use loads),
//      apply 4x4 mix, NT write output
// ---------------------------------------------------------------------------
__global__ __launch_bounds__(256)
void mchc_fused_kernel(const float* __restrict__ x,
                       const v16bf* __restrict__ bfrag, // packed W_router
                       const float* __restrict__ br,    // (24,)
                       float* __restrict__ out)
{
    __shared__ __align__(16) unsigned short routing[TOK_TILE][PITCH];
    __shared__ float raw_s[TOK_TILE][NOUT];
    __shared__ float Mc[TOK_TILE][16];

    const int tid   = threadIdx.x;
    const int tbase = blockIdx.x * TOK_TILE;

    v8f acc0 = {};   // N = 0..15
    v8f acc1 = {};   // N = 16..31 (24..31 are zero columns)

    const int lane = tid & 31;
    const int g    = lane >> 4;
    const int nc   = lane & 15;

    for (int half = 0; half < 2; ++half) {
        // ---- Phase 1: means -> LDS bf16 -----------------------------------
        for (int it = 0; it < 16; ++it) {
            int item = it * 256 + tid;
            int tt   = item >> 8;
            int db   = item & 255;
            int tok  = tbase + tt;
            int d    = half * D_HALF + db * 4;

            // prefetch 2 iterations ahead (global_prefetch_b8, HBM first touch)
            if (it + 2 < 16) {
                int ni  = item + 512;
                int ntt = ni >> 8, ndb = ni & 255;
                int ntok = tbase + ntt;
                int nd   = half * D_HALF + ndb * 4;
                #pragma unroll
                for (int j = 0; j < 4; ++j)
                    __builtin_prefetch(x + ((ntok * WS + j) * DD + nd), 0, 3);
            }

            v4f a = *(const v4f*)(x + ((tok * WS + 0) * DD + d));
            v4f b = *(const v4f*)(x + ((tok * WS + 1) * DD + d));
            v4f c = *(const v4f*)(x + ((tok * WS + 2) * DD + d));
            v4f e = *(const v4f*)(x + ((tok * WS + 3) * DD + d));
            int dl = db * 4;
            routing[tt][dl + 0] = f32_to_bf16_bits(0.25f * (a.x + b.x + c.x + e.x));
            routing[tt][dl + 1] = f32_to_bf16_bits(0.25f * (a.y + b.y + c.y + e.y));
            routing[tt][dl + 2] = f32_to_bf16_bits(0.25f * (a.z + b.z + c.z + e.z));
            routing[tt][dl + 3] = f32_to_bf16_bits(0.25f * (a.w + b.w + c.w + e.w));
        }
        __syncthreads();

        // ---- Phase 2: WMMA router matmul (wave 0, EXEC all-ones) ----------
        if (tid < 32) {
            for (int kc = 0; kc < D_HALF / 32; ++kc) {
                int klocal = kc * 32;
                int kcg    = half * (D_HALF / 32) + kc;
                // A fragment: two contiguous 16B runs from LDS
                v8bf a0 = *(const v8bf*)&routing[nc][klocal + g * 8];
                v8bf a1 = *(const v8bf*)&routing[nc][klocal + 16 + g * 8];
                v16bf afrag = __builtin_shufflevector(
                    a0, a1, 0, 1, 2, 3, 4, 5, 6, 7, 8, 9, 10, 11, 12, 13, 14, 15);
                // B fragments: pre-packed, contiguous 32B loads
                v16bf b0 = bfrag[(kcg * 2 + 0) * 32 + lane];
                v16bf b1 = bfrag[(kcg * 2 + 1) * 32 + lane];
                acc0 = __builtin_amdgcn_wmma_f32_16x16x32_bf16(
                    false, afrag, false, b0, (short)0, acc0, false, false);
                acc1 = __builtin_amdgcn_wmma_f32_16x16x32_bf16(
                    false, afrag, false, b1, (short)0, acc1, false, false);
            }
        }
        __syncthreads();  // before next half overwrites `routing`
    }

    // ---- Writeback logits (C/D layout: row = r + 8g, col = nc) ------------
    if (tid < 32) {
        float bb0 = br[nc];
        int   n1  = 16 + nc;
        float bb1 = (n1 < NOUT) ? br[n1] : 0.0f;
        #pragma unroll
        for (int r = 0; r < 8; ++r) {
            int M = r + 8 * g;
            raw_s[M][nc] = acc0[r] + bb0;
            if (n1 < NOUT) raw_s[M][n1] = acc1[r] + bb1;
        }
    }
    __syncthreads();

    // ---- Phase 3: softmaxes + Sinkhorn -> fused 4x4 coefficients ----------
    if (tid < TOK_TILE) {
        float r[NOUT];
        #pragma unroll
        for (int k = 0; k < NOUT; ++k) r[k] = raw_s[tid][k];

        float pre[4], post[4];
        {
            float m = fmaxf(fmaxf(r[0], r[1]), fmaxf(r[2], r[3]));
            float s = 0.f;
            #pragma unroll
            for (int i = 0; i < 4; ++i) { pre[i] = __expf(r[i] - m); s += pre[i]; }
            float inv = 1.0f / s;
            #pragma unroll
            for (int i = 0; i < 4; ++i) pre[i] *= inv;
        }
        {
            float m = fmaxf(fmaxf(r[4], r[5]), fmaxf(r[6], r[7]));
            float s = 0.f;
            #pragma unroll
            for (int i = 0; i < 4; ++i) { post[i] = __expf(r[4 + i] - m); s += post[i]; }
            float inv = 1.0f / s;
            #pragma unroll
            for (int i = 0; i < 4; ++i) post[i] *= inv;
        }
        float w[4][4];
        #pragma unroll
        for (int i = 0; i < 4; ++i)
            #pragma unroll
            for (int j = 0; j < 4; ++j)
                w[i][j] = __expf(r[8 + i * 4 + j] + ((i == j) ? 2.0f : -2.0f));
        #pragma unroll
        for (int itc = 0; itc < 4; ++itc) {
            #pragma unroll
            for (int i = 0; i < 4; ++i) {
                float rs = fmaxf(w[i][0] + w[i][1] + w[i][2] + w[i][3], 1e-6f);
                float inv = 1.0f / rs;
                #pragma unroll
                for (int j = 0; j < 4; ++j) w[i][j] *= inv;
            }
            #pragma unroll
            for (int j = 0; j < 4; ++j) {
                float cs = fmaxf(w[0][j] + w[1][j] + w[2][j] + w[3][j], 1e-6f);
                float inv = 1.0f / cs;
                #pragma unroll
                for (int i = 0; i < 4; ++i) w[i][j] *= inv;
            }
        }
        #pragma unroll
        for (int i = 0; i < 4; ++i)
            #pragma unroll
            for (int j = 0; j < 4; ++j)
                Mc[tid][i * 4 + j] = w[i][j] + post[i] * pre[j];
    }
    __syncthreads();

    // ---- Phase 4: stream x (reverse order, NT/last-use), mix, NT store ----
    for (int it = 31; it >= 0; --it) {
        int item = it * 256 + tid;
        int tt   = item >> 9;
        int dblk = item & 511;
        int tok  = tbase + tt;
        int d    = dblk * 4;

        v4f xs[4];
        #pragma unroll
        for (int j = 0; j < 4; ++j)
            xs[j] = __builtin_nontemporal_load(
                        (const v4f*)(x + ((tok * WS + j) * DD + d)));

        #pragma unroll
        for (int i = 0; i < 4; ++i) {
            float c0 = Mc[tt][i * 4 + 0], c1 = Mc[tt][i * 4 + 1];
            float c2 = Mc[tt][i * 4 + 2], c3 = Mc[tt][i * 4 + 3];
            v4f o;
            o.x = fmaf(c0, xs[0].x, fmaf(c1, xs[1].x, fmaf(c2, xs[2].x, c3 * xs[3].x)));
            o.y = fmaf(c0, xs[0].y, fmaf(c1, xs[1].y, fmaf(c2, xs[2].y, c3 * xs[3].y)));
            o.z = fmaf(c0, xs[0].z, fmaf(c1, xs[1].z, fmaf(c2, xs[2].z, c3 * xs[3].z)));
            o.w = fmaf(c0, xs[0].w, fmaf(c1, xs[1].w, fmaf(c2, xs[2].w, c3 * xs[3].w)));
            __builtin_nontemporal_store(o, (v4f*)(out + ((tok * WS + i) * DD + d)));
        }
    }
}

extern "C" void kernel_launch(void* const* d_in, const int* in_sizes, int n_in,
                              void* d_out, int out_size, void* d_ws, size_t ws_size,
                              hipStream_t stream) {
    (void)in_sizes; (void)n_in; (void)out_size; (void)ws_size;
    const float* x  = (const float*)d_in[0];
    const float* Wr = (const float*)d_in[1];
    const float* br = (const float*)d_in[2];
    float* out = (float*)d_out;
    v16bf* bfrag = (v16bf*)d_ws;   // 64 chunks * 2 tiles * 32 lanes * 32B = 128 KB

    // Stage 1: pack router weights into WMMA B-fragment layout (L2-resident)
    pack_router_kernel<<<dim3(NCHUNK), dim3(64), 0, stream>>>(Wr, bfrag);
    // Stage 2: fused router + Sinkhorn + mix
    mchc_fused_kernel<<<dim3(BT / TOK_TILE), dim3(256), 0, stream>>>(x, bfrag, br, out);
}
